// RETTransformerBlock_42073499632184
// MI455X (gfx1250) — matrix operand order repaired and checked by reading, not compiled, not run on hardware
//
#include <hip/hip_runtime.h>

// ---------------------------------------------------------------------------
// Transformer block on MI455X (gfx1250): bf16 WMMA GEMMs + flash attention.
// B=2, S=2048, D=1024, H=16, HD=64.
// ---------------------------------------------------------------------------

#define B_SZ   2
#define S_LEN  2048
#define D_DIM  1024
#define NHEAD  16
#define HDIM   64
#define NROWS  (B_SZ * S_LEN)   // 4096

typedef __attribute__((ext_vector_type(16))) __bf16 v16bf;
typedef __attribute__((ext_vector_type(8)))  float  v8f;

union FragU {
    unsigned int u[8];
    v16bf v;
};

__device__ __forceinline__ v8f wmma_bf16(v16bf a, v16bf b, v8f c) {
    return __builtin_amdgcn_wmma_f32_16x16x32_bf16(
        /*neg_a=*/false, a, /*neg_b=*/false, b,
        /*c_mod=*/(short)0, c, /*reuse_a=*/false, /*reuse_b=*/false);
}

// A fragment (16x32, M x K): lane holds row M = row param; per VGPR v the K
// offset is (v/4)*16 + (lane/16)*8 + (v&3)*2 per ISA 7.12.2. K pairs are
// contiguous in memory -> 32-bit loads the compiler merges into ds_load_b128.
__device__ __forceinline__ v16bf load_frag_a(const __bf16* base, int row_stride,
                                             int row, int lane) {
    FragU f;
    const __bf16* p = base + row * row_stride;
#pragma unroll
    for (int v = 0; v < 8; ++v) {
        int kk = ((v >> 2) << 4) + ((lane >> 4) << 3) + ((v & 3) << 1);
        f.u[v] = *reinterpret_cast<const unsigned int*>(p + kk);
    }
    return f.v;
}

// B fragment (32x16, K x N) stored as rows of N with contiguous K:
// lane holds column N = col_row; K offset = (v/4)*16 + (lane/16)*8 + (v&3)*2.
__device__ __forceinline__ v16bf load_frag_b(const __bf16* base, int row_stride,
                                             int col_row, int lane) {
    FragU f;
    const __bf16* p = base + col_row * row_stride;
#pragma unroll
    for (int v = 0; v < 8; ++v) {
        int kk = ((v >> 2) << 4) + ((lane >> 4) << 3) + ((v & 3) << 1);
        f.u[v] = *reinterpret_cast<const unsigned int*>(p + kk);
    }
    return f.v;
}

// gfx1250 async copy: LDS[lds_off..+15] = MEM[sbase + voff .. +15], ASYNCcnt.
// GVS mode: uniform SGPR base + per-lane signed 32-bit byte offset.
__device__ __forceinline__ void async_copy_b128(unsigned lds_off, unsigned voff,
                                                const void* sbase) {
    asm volatile("global_load_async_to_lds_b128 %0, %1, %2 offset:0"
                 :: "v"(lds_off), "v"(voff), "s"(sbase)
                 : "memory");
}
__device__ __forceinline__ void wait_asynccnt0() {
    asm volatile("s_wait_asynccnt 0x0" ::: "memory");
}

// ---------------------------------------------------------------------------
// fp32 -> bf16 elementwise convert (weights)
// ---------------------------------------------------------------------------
__global__ __launch_bounds__(256) void cvt_bf16_kernel(const float* __restrict__ src,
                                                       __bf16* __restrict__ dst, int n) {
    int i = (blockIdx.x * 256 + threadIdx.x) * 4;
    if (i + 3 < n) {
        float4 v = *reinterpret_cast<const float4*>(src + i);
        dst[i + 0] = (__bf16)v.x;
        dst[i + 1] = (__bf16)v.y;
        dst[i + 2] = (__bf16)v.z;
        dst[i + 3] = (__bf16)v.w;
    }
}

// ---------------------------------------------------------------------------
// RMSNorm: one 1024-wide row per 256-thread block, bf16 output.
// ---------------------------------------------------------------------------
__global__ __launch_bounds__(256) void rmsnorm_kernel(const float* __restrict__ x,
                                                      const float* __restrict__ w,
                                                      __bf16* __restrict__ out) {
    __shared__ float red[8];
    const int row = blockIdx.x;
    const int t = threadIdx.x;
    float4 xv = reinterpret_cast<const float4*>(x + row * D_DIM)[t];
    float ss = xv.x * xv.x + xv.y * xv.y + xv.z * xv.z + xv.w * xv.w;
#pragma unroll
    for (int off = 16; off > 0; off >>= 1) ss += __shfl_xor(ss, off, 32);
    if ((t & 31) == 0) red[t >> 5] = ss;
    __syncthreads();
    float tot = 0.f;
#pragma unroll
    for (int i = 0; i < 8; ++i) tot += red[i];
    const float r = rsqrtf(tot * (1.0f / (float)D_DIM) + 1e-6f);
    float4 wv = reinterpret_cast<const float4*>(w)[t];
    __bf16* o = out + row * D_DIM + t * 4;
    o[0] = (__bf16)(xv.x * r * wv.x);
    o[1] = (__bf16)(xv.y * r * wv.y);
    o[2] = (__bf16)(xv.z * r * wv.z);
    o[3] = (__bf16)(xv.w * r * wv.w);
}

// ---------------------------------------------------------------------------
// GEMM: out[M,N] = A[M,K] @ W[N,K]^T  (bf16 inputs, fp32 accumulate)
// Workgroup tile 128x128, 8 waves each computing 64(M)x32(N), K-step 64.
// Tiles staged with global_load_async_to_lds_b128 (ASYNCcnt path, no VGPR
// round-trip). LDS rows padded to 72 elements (144B): 16B-aligned b128
// transfers, 36-bank row stride -> conflict-free 16-lane fragment gathers.
// ---------------------------------------------------------------------------
template <bool RESID, bool OUTBF16>
__global__ __launch_bounds__(256)
void gemm_bf16_kernel(const __bf16* __restrict__ A, const __bf16* __restrict__ W,
                      const float* __restrict__ resid, void* __restrict__ outp,
                      int M, int N, int K) {
    __shared__ __bf16 lA[128 * 72];
    __shared__ __bf16 lB[128 * 72];

    const int lane   = threadIdx.x & 31;
    const int wave   = threadIdx.x >> 5;
    const int waveM  = (wave & 1) * 64;
    const int waveN  = (wave >> 1) * 32;
    const int blockN = blockIdx.x * 128;
    const int blockM = blockIdx.y * 128;
    const int cprow  = threadIdx.x >> 1;        // 0..127 copy row
    const int cphalf = (threadIdx.x & 1) * 32;  // 0 or 32 (elements)

    const unsigned ldsA0 = (unsigned)(size_t)&lA[cprow * 72 + cphalf];
    const unsigned ldsB0 = (unsigned)(size_t)&lB[cprow * 72 + cphalf];

    v8f acc[4][2];
#pragma unroll
    for (int mb = 0; mb < 4; ++mb)
#pragma unroll
        for (int nb = 0; nb < 2; ++nb) acc[mb][nb] = (v8f){};

    for (int k0 = 0; k0 < K; k0 += 64) {
        __syncthreads();
        // Stage 128x64 A and W tiles: 4 async b128 per thread per tile.
        unsigned aoff = (unsigned)(((blockM + cprow) * K + k0 + cphalf) * 2);
        unsigned boff = (unsigned)(((blockN + cprow) * K + k0 + cphalf) * 2);
#pragma unroll
        for (int c = 0; c < 4; ++c) {
            async_copy_b128(ldsA0 + c * 16, aoff + c * 16, A);
            async_copy_b128(ldsB0 + c * 16, boff + c * 16, W);
        }
        wait_asynccnt0();
        __syncthreads();

        v16bf af[2][4], bfr[2][2];
#pragma unroll
        for (int kh = 0; kh < 2; ++kh) {
#pragma unroll
            for (int mb = 0; mb < 4; ++mb)
                af[kh][mb] =
                    load_frag_a(lA + kh * 32, 72, waveM + mb * 16 + (lane & 15), lane);
#pragma unroll
            for (int nb = 0; nb < 2; ++nb)
                bfr[kh][nb] =
                    load_frag_b(lB + kh * 32, 72, waveN + nb * 16 + (lane & 15), lane);
        }
#pragma unroll
        for (int kh = 0; kh < 2; ++kh)
#pragma unroll
            for (int mb = 0; mb < 4; ++mb)
#pragma unroll
                for (int nb = 0; nb < 2; ++nb)
                    acc[mb][nb] = wmma_bf16(af[kh][mb], bfr[kh][nb], acc[mb][nb]);
    }

#pragma unroll
    for (int mb = 0; mb < 4; ++mb)
#pragma unroll
        for (int nb = 0; nb < 2; ++nb)
#pragma unroll
            for (int r = 0; r < 8; ++r) {
                int gM = blockM + waveM + mb * 16 + ((lane >> 4) << 3) + r;
                int gN = blockN + waveN + nb * 16 + (lane & 15);
                float val = acc[mb][nb][r];
                if constexpr (RESID) val += resid[gM * N + gN];
                if constexpr (OUTBF16)
                    reinterpret_cast<__bf16*>(outp)[gM * N + gN] = (__bf16)val;
                else
                    reinterpret_cast<float*>(outp)[gM * N + gN] = val;
            }
}

// ---------------------------------------------------------------------------
// Flash attention: grid (S/64, B*H), 128 threads (4 waves). Each wave owns a
// 16-query tile; block cooperatively stages 32-key K (row-major) and V
// (transposed) tiles in LDS. Online softmax via 16-lane shuffle reductions
// (matches C-matrix row layout: VGPR r <-> rows r / r+8 per lane half).
// ---------------------------------------------------------------------------
__global__ __launch_bounds__(128)
void attn_kernel(const __bf16* __restrict__ qg, const __bf16* __restrict__ kg,
                 const __bf16* __restrict__ vg, __bf16* __restrict__ ctx) {
    __shared__ __bf16 lK[32 * 72];    // [key][hd], row pad 64->72
    __shared__ __bf16 lVt[64 * 40];   // [hd][key], row pad 32->40
    __shared__ __bf16 lP[4][16 * 40]; // per-wave P scratch (C->A relayout)

    const int lane = threadIdx.x & 31;
    const int wave = threadIdx.x >> 5;
    const int bh = blockIdx.y;
    const int b = bh >> 4, h = bh & 15;
    const int rowBase = b * S_LEN;      // row offset in [4096, 1024]
    const int colBase = h * HDIM;       // column offset
    const int qt = blockIdx.x * 4 + wave;

    // Q fragments, loaded once from global (K pairs contiguous over hd).
    const __bf16* qbase = qg + (rowBase + qt * 16) * D_DIM + colBase;
    v16bf qf[2];
    qf[0] = load_frag_a(qbase + 0,  D_DIM, lane & 15, lane);
    qf[1] = load_frag_a(qbase + 32, D_DIM, lane & 15, lane);

    float mrun[8], lrun[8];
#pragma unroll
    for (int r = 0; r < 8; ++r) { mrun[r] = -1e30f; lrun[r] = 0.f; }
    v8f o0 = (v8f){}, o1 = (v8f){}, o2 = (v8f){}, o3 = (v8f){};

    const int key = threadIdx.x >> 2;         // 0..31
    const int seg = (threadIdx.x & 3) * 16;   // 0,16,32,48

    for (int kb = 0; kb < S_LEN; kb += 32) {
        __syncthreads();
        {   // K tile: row-major [key][hd]
            const __bf16* s = kg + (rowBase + kb + key) * D_DIM + colBase + seg;
            uint4* d = reinterpret_cast<uint4*>(&lK[key * 72 + seg]);
            d[0] = reinterpret_cast<const uint4*>(s)[0];
            d[1] = reinterpret_cast<const uint4*>(s)[1];
        }
        {   // V tile: transposed [hd][key]
            const __bf16* s = vg + (rowBase + kb + key) * D_DIM + colBase + seg;
            __bf16 tmp[16];
            *reinterpret_cast<uint4*>(&tmp[0]) = reinterpret_cast<const uint4*>(s)[0];
            *reinterpret_cast<uint4*>(&tmp[8]) = reinterpret_cast<const uint4*>(s)[1];
#pragma unroll
            for (int i = 0; i < 16; ++i) lVt[(seg + i) * 40 + key] = tmp[i];
        }
        __syncthreads();

        // scores: S[16q x 32k] = Q(16x64) @ K_tile(32x64)^T, two hd steps
        v8f s0 = (v8f){}, s1 = (v8f){};
#pragma unroll
        for (int f = 0; f < 2; ++f) {
            v16bf bk0 = load_frag_b(lK + f * 32, 72, (lane & 15), lane);
            v16bf bk1 = load_frag_b(lK + f * 32, 72, 16 + (lane & 15), lane);
            s0 = wmma_bf16(qf[f], bk0, s0);
            s1 = wmma_bf16(qf[f], bk1, s1);
        }

        __bf16* lPw = lP[wave];
        const float sc = 0.125f;  // 1/sqrt(64)
#pragma unroll
        for (int r = 0; r < 8; ++r) {
            float e0 = s0[r] * sc, e1 = s1[r] * sc;
            float mx = fmaxf(e0, e1);
            mx = fmaxf(mx, __shfl_xor(mx, 1, 16));
            mx = fmaxf(mx, __shfl_xor(mx, 2, 16));
            mx = fmaxf(mx, __shfl_xor(mx, 4, 16));
            mx = fmaxf(mx, __shfl_xor(mx, 8, 16));
            float mn = fmaxf(mrun[r], mx);
            float alpha = __expf(mrun[r] - mn);
            e0 = __expf(e0 - mn);
            e1 = __expf(e1 - mn);
            float sm = e0 + e1;
            sm += __shfl_xor(sm, 1, 16);
            sm += __shfl_xor(sm, 2, 16);
            sm += __shfl_xor(sm, 4, 16);
            sm += __shfl_xor(sm, 8, 16);
            lrun[r] = lrun[r] * alpha + sm;
            mrun[r] = mn;
            o0[r] *= alpha; o1[r] *= alpha; o2[r] *= alpha; o3[r] *= alpha;
            int prow = ((lane >> 4) << 3) + r;
            lPw[prow * 40 + (lane & 15)]      = (__bf16)e0;
            lPw[prow * 40 + 16 + (lane & 15)] = (__bf16)e1;
        }

        // O += P(16x32) @ V_tile(32x64)
        v16bf pf = load_frag_a(lPw, 40, lane & 15, lane);
        o0 = wmma_bf16(pf, load_frag_b(lVt, 40,  0 + (lane & 15), lane), o0);
        o1 = wmma_bf16(pf, load_frag_b(lVt, 40, 16 + (lane & 15), lane), o1);
        o2 = wmma_bf16(pf, load_frag_b(lVt, 40, 32 + (lane & 15), lane), o2);
        o3 = wmma_bf16(pf, load_frag_b(lVt, 40, 48 + (lane & 15), lane), o3);
    }

#pragma unroll
    for (int r = 0; r < 8; ++r) {
        float inv = 1.0f / lrun[r];
        int gr = rowBase + qt * 16 + ((lane >> 4) << 3) + r;
        __bf16* dst = ctx + gr * D_DIM + colBase + (lane & 15);
        dst[0]  = (__bf16)(o0[r] * inv);
        dst[16] = (__bf16)(o1[r] * inv);
        dst[32] = (__bf16)(o2[r] * inv);
        dst[48] = (__bf16)(o3[r] * inv);
    }
}

// ---------------------------------------------------------------------------
// a = silu(g) * u  -> bf16
// ---------------------------------------------------------------------------
__global__ __launch_bounds__(256) void silu_mul_kernel(const float* __restrict__ g,
                                                       const float* __restrict__ u,
                                                       __bf16* __restrict__ a, int n) {
    int i = (blockIdx.x * 256 + threadIdx.x) * 4;
    if (i + 3 < n) {
        float4 gv = *reinterpret_cast<const float4*>(g + i);
        float4 uv = *reinterpret_cast<const float4*>(u + i);
        a[i + 0] = (__bf16)((gv.x / (1.f + __expf(-gv.x))) * uv.x);
        a[i + 1] = (__bf16)((gv.y / (1.f + __expf(-gv.y))) * uv.y);
        a[i + 2] = (__bf16)((gv.z / (1.f + __expf(-gv.z))) * uv.z);
        a[i + 3] = (__bf16)((gv.w / (1.f + __expf(-gv.w))) * uv.w);
    }
}

// ---------------------------------------------------------------------------
// Host launch
// ---------------------------------------------------------------------------
extern "C" void kernel_launch(void* const* d_in, const int* in_sizes, int n_in,
                              void* d_out, int out_size, void* d_ws, size_t ws_size,
                              hipStream_t stream) {
    (void)in_sizes; (void)n_in; (void)out_size; (void)ws_size;

    const float* x    = (const float*)d_in[0];
    const float* w_in[7] = {(const float*)d_in[1], (const float*)d_in[2],
                            (const float*)d_in[3], (const float*)d_in[4],
                            (const float*)d_in[5], (const float*)d_in[6],
                            (const float*)d_in[7]};
    const float* ln1 = (const float*)d_in[8];
    const float* ln2 = (const float*)d_in[9];
    float* out = (float*)d_out;

    // Workspace layout (bytes). Lifetimes allow aliasing g/u over q..ctx.
    const size_t MB = (size_t)1 << 20;
    char* ws = (char*)d_ws;
    __bf16* wb[7];
    for (int i = 0; i < 7; ++i) wb[i] = (__bf16*)(ws + (size_t)i * 2 * MB);  // 0..14MB
    __bf16* hbuf   = (__bf16*)(ws + 14 * MB);   // 8MB  (h / h2 / a)
    __bf16* qbuf   = (__bf16*)(ws + 22 * MB);   // 8MB
    __bf16* kbuf   = (__bf16*)(ws + 30 * MB);   // 8MB
    __bf16* vbuf   = (__bf16*)(ws + 38 * MB);   // 8MB
    __bf16* ctxbuf = (__bf16*)(ws + 46 * MB);   // 8MB
    float*  x1     = (float*)(ws + 54 * MB);    // 16MB
    float*  gbuf   = (float*)(ws + 22 * MB);    // 16MB (aliases q+k, dead by then)
    float*  ubuf   = (float*)(ws + 38 * MB);    // 16MB (aliases v+ctx, dead by then)
    __bf16* abuf   = hbuf;                      // aliases h2 after gate/up GEMMs

    const int NW = D_DIM * D_DIM;               // 1M elements per weight
    const dim3 gemm_grid(D_DIM / 128, NROWS / 128);  // (8, 32)

    // 1) weights -> bf16
    for (int i = 0; i < 7; ++i)
        cvt_bf16_kernel<<<NW / (256 * 4), 256, 0, stream>>>(w_in[i], wb[i], NW);

    // 2) h = rmsnorm(x, ln1)
    rmsnorm_kernel<<<NROWS, 256, 0, stream>>>(x, ln1, hbuf);

    // 3) q/k/v = h @ W^T  (bf16 out)
    gemm_bf16_kernel<false, true><<<gemm_grid, 256, 0, stream>>>(
        hbuf, wb[0], nullptr, qbuf, NROWS, D_DIM, D_DIM);
    gemm_bf16_kernel<false, true><<<gemm_grid, 256, 0, stream>>>(
        hbuf, wb[1], nullptr, kbuf, NROWS, D_DIM, D_DIM);
    gemm_bf16_kernel<false, true><<<gemm_grid, 256, 0, stream>>>(
        hbuf, wb[2], nullptr, vbuf, NROWS, D_DIM, D_DIM);

    // 4) ctx = flash_attention(q, k, v)
    attn_kernel<<<dim3(S_LEN / 64, B_SZ * NHEAD), 128, 0, stream>>>(
        qbuf, kbuf, vbuf, ctxbuf);

    // 5) x1 = x + ctx @ wo^T  (fp32 out + residual)
    gemm_bf16_kernel<true, false><<<gemm_grid, 256, 0, stream>>>(
        ctxbuf, wb[3], x, x1, NROWS, D_DIM, D_DIM);

    // 6) h2 = rmsnorm(x1, ln2)
    rmsnorm_kernel<<<NROWS, 256, 0, stream>>>(x1, ln2, hbuf);

    // 7) g = h2 @ wg^T, u = h2 @ wu^T  (fp32 out)
    gemm_bf16_kernel<false, false><<<gemm_grid, 256, 0, stream>>>(
        hbuf, wb[4], nullptr, gbuf, NROWS, D_DIM, D_DIM);
    gemm_bf16_kernel<false, false><<<gemm_grid, 256, 0, stream>>>(
        hbuf, wb[5], nullptr, ubuf, NROWS, D_DIM, D_DIM);

    // 7.5) a = silu(g) * u  (bf16)
    silu_mul_kernel<<<(NROWS * D_DIM) / (256 * 4), 256, 0, stream>>>(
        gbuf, ubuf, abuf, NROWS * D_DIM);

    // 8) out = x1 + a @ wd^T
    gemm_bf16_kernel<true, false><<<gemm_grid, 256, 0, stream>>>(
        abuf, wb[6], x1, out, NROWS, D_DIM, D_DIM);
}
